// MaskGAE_stage1_25615184953521
// MI455X (gfx1250) — matrix operand order repaired and checked by reading, not compiled, not run on hardware
//
#include <hip/hip_runtime.h>
#include <stdint.h>

typedef __attribute__((ext_vector_type(16))) _Float16 v16h;
typedef __attribute__((ext_vector_type(8)))  _Float16 v8h;
typedef __attribute__((ext_vector_type(8)))  float    v8f;
typedef __attribute__((ext_vector_type(4)))  unsigned int u32x4;
typedef __attribute__((ext_vector_type(8)))  int i32x8;
typedef __attribute__((ext_vector_type(4)))  int i32x4;

#define BN_EPS 1e-5f
#define LDK 136   // 128 halves + 8 halves pad -> 272B row stride, conflict-free ds_load_b128

#if __has_builtin(__builtin_amdgcn_tensor_load_to_lds)
#define USE_TDM 1
#else
#define USE_TDM 0
#endif

enum { STAGE_EDGE_CAT = 0, STAGE_NODE_CAT = 1, STAGE_PLAIN_H = 2 };

// C[M, NOUT] = epilogue( A_tile[M,128] @ Wh[NOUT,128]^T )
// A staged into LDS as f16 (manual gather/concat, or TDM for f16 inputs).
// MTILES 16-row tiles per block; one 16x16 WMMA tile per wave per M-tile,
// B fragment loaded once per k-step and reused across M-tiles.
template<int MODE, int NOUT, int MTILES, bool DO_BN, bool DO_ATOMIC, bool DO_STORE, bool F16OUT>
__launch_bounds__(256)
__global__ void gemm_wmma_kernel(
    const void*  __restrict__ A,        // STAGE_PLAIN_H: f16 input [M,128]
    const float* __restrict__ n1,       // [*,64] node half-features
    const float* __restrict__ aggOrEa,  // node: agg [N,64]; edge: edge_attr [E]
    const float* __restrict__ We0,      // edge: l0_We column [64]
    const int*   __restrict__ srcIdx,   // [E]
    const int*   __restrict__ dstIdx,   // [E]
    const _Float16* __restrict__ Wh,    // [NOUT,128] f16 weights (B = W^T)
    const float* __restrict__ bias,     // [NOUT] or nullptr
    const float* __restrict__ bnG, const float* __restrict__ bnB,
    const float* __restrict__ bnM, const float* __restrict__ bnV,
    void*  __restrict__ out,            // [M,NOUT] f32 or f16
    float* __restrict__ aggOut,         // [N,64] atomic-accumulate target
    int M)
{
    constexpr int ROWS = 16 * MTILES;
    __shared__ __align__(16) _Float16 lds[ROWS * LDK];
    __shared__ int s_src[ROWS], s_dst[ROWS];
    const int mbase = blockIdx.x * ROWS;

    // ---- stage ROWS x 128 A tile into LDS as f16 ----
    if (MODE == STAGE_EDGE_CAT) {
        if (threadIdx.x < ROWS) {
            const int grow = mbase + threadIdx.x;
            const bool ok = grow < M;
            s_src[threadIdx.x] = ok ? srcIdx[grow] : 0;
            s_dst[threadIdx.x] = ok ? dstIdx[grow] : 0;
        }
        __syncthreads();
        for (int idx = threadIdx.x; idx < ROWS * 128; idx += NOUT * 2) {
            const int r = idx >> 7, k = idx & 127;
            const int grow = mbase + r;
            float v = 0.f;
            if (grow < M) {
                if (k < 64) {
                    v = n1[s_src[r] * 64 + k] + n1[s_dst[r] * 64 + k];
                } else {
                    float t = aggOrEa[grow] * We0[k - 64];   // e1 = relu(ea*We) on the fly
                    v = t > 0.f ? t : 0.f;
                }
            }
            lds[r * LDK + k] = (_Float16)v;
        }
    } else if (MODE == STAGE_NODE_CAT) {
        for (int idx = threadIdx.x; idx < ROWS * 128; idx += NOUT * 2) {
            const int r = idx >> 7, k = idx & 127;
            const int grow = mbase + r;
            float v = 0.f;
            if (grow < M)
                v = (k < 64) ? aggOrEa[grow * 64 + k] : n1[grow * 64 + (k - 64)];
            lds[r * LDK + k] = (_Float16)v;
        }
    } else {
        const _Float16* Ah = (const _Float16*)A;
#if USE_TDM
        // One wave drives the Tensor Data Mover: 2D tile ROWSx128 f16, LDS pad
        // 4 DWORDs every 64 DWORDs -> 272B padded row stride (matches LDK).
        if (threadIdx.x < 32) {
            const unsigned long long ga =
                (unsigned long long)(uintptr_t)(Ah + (size_t)mbase * 128);
            const unsigned ldsa = (unsigned)(uintptr_t)(&lds[0]); // low 32b = LDS offset
            u32x4 g0;
            g0.x = 1u;                                   // count=1 (valid descriptor)
            g0.y = ldsa;                                 // lds_addr
            g0.z = (unsigned)(ga & 0xFFFFFFFFu);         // global_addr[31:0]
            g0.w = (unsigned)((ga >> 32) & 0x01FFFFFFu)  // global_addr[56:32]
                   | (2u << 30);                         // type=2 (image)
            const unsigned drem = (unsigned)(M - mbase); // tensor_dim1 (OOB rows -> 0)
            i32x8 g1 = {
                (int)((1u << 16) | (1u << 20) | (5u << 22) | (3u << 25)),
                //     data_size=2B  pad_en     ival=64DW     amt=4DW
                (int)(128u << 16),                       // tensor_dim0 = 128
                (int)((drem & 0xFFFFu) << 16),           // tensor_dim1[15:0]
                (int)((drem >> 16) & 0xFFFFu) | (int)(128u << 16), // dim1[31:16] | tile_dim0
                (int)ROWS,                               // tile_dim1
                128,                                     // tensor_dim0_stride
                0, 0
            };
            i32x4 gz = {0, 0, 0, 0};
#if defined(__clang_major__) && (__clang_major__ >= 23)
            i32x8 gz8 = {0, 0, 0, 0, 0, 0, 0, 0};
            __builtin_amdgcn_tensor_load_to_lds(g0, g1, gz, gz, gz8, 0);
#else
            __builtin_amdgcn_tensor_load_to_lds(g0, g1, gz, gz, 0);
#endif
            __builtin_amdgcn_s_wait_tensorcnt(0);
        }
#else
        for (int idx = threadIdx.x; idx < ROWS * 128; idx += NOUT * 2) {
            const int r = idx >> 7, k = idx & 127;
            const int grow = mbase + r;
            lds[r * LDK + k] = (grow < M) ? Ah[(size_t)grow * 128 + k] : (_Float16)0.f;
        }
#endif
    }
    __syncthreads();

    // ---- WMMA: wave w owns output columns [16w,16w+16) for all MTILES row tiles ----
    const int wave = threadIdx.x >> 5;
    const int lane = threadIdx.x & 31;
    const int ntile = wave * 16;
    const int l16 = lane & 15;
    const int lh  = lane >> 4;

    v8f c[MTILES] = {};
    const _Float16* brow = &Wh[(ntile + l16) * 128 + lh * 16];
    #pragma unroll
    for (int kt = 0; kt < 4; ++kt) {
        // B 32x16: lane = N col; K = kt*32 + lh*16 + elem (contiguous in W row)
        v16h b = *(const v16h*)(brow + kt * 32);
        #pragma unroll
        for (int mt = 0; mt < MTILES; ++mt) {
            // A 16x32: lane = M row; K = kt*32 + lh*16 + {0..15}
            const _Float16* arow = &lds[(mt * 16 + l16) * LDK + kt * 32 + lh * 16];
            v8h alo = *(const v8h*)(arow);
            v8h ahi = *(const v8h*)(arow + 8);
            v16h a = __builtin_shufflevector(alo, ahi,
                                             0,1,2,3,4,5,6,7,8,9,10,11,12,13,14,15);
            c[mt] = __builtin_amdgcn_wmma_f32_16x16x32_f16(false, a, false, b,
                                                           (short)0, c[mt], false, false);
        }
    }

    // ---- epilogue: bias -> relu -> optional BN; f32/f16 store, fused segment-sum ----
    const int col = ntile + l16;
    const float bi = bias ? bias[col] : 0.f;
    float gg = 0.f, bb = 0.f, mm = 0.f, iv = 0.f;
    if (DO_BN) {
        gg = bnG[col]; bb = bnB[col]; mm = bnM[col];
        iv = rsqrtf(bnV[col] + BN_EPS);
    }
    float*     outF = (float*)out;
    _Float16*  outH = (_Float16*)out;
    #pragma unroll
    for (int mt = 0; mt < MTILES; ++mt) {
        #pragma unroll
        for (int v = 0; v < 8; ++v) {
            const int grow = mbase + mt * 16 + v + lh * 8; // C layout: M = v + 8*(lane>=16)
            if (grow < M) {
                float y = c[mt][v] + bi;
                y = y > 0.f ? y : 0.f;
                if (DO_BN) y = (y - mm) * iv * gg + bb;
                if (DO_STORE) {
                    if (F16OUT) outH[(size_t)grow * NOUT + col] = (_Float16)y;
                    else        outF[(size_t)grow * NOUT + col] = y;
                }
                if (DO_ATOMIC)
                    atomicAdd(&aggOut[(size_t)dstIdx[grow] * 64 + col], y);
            }
        }
    }
}

__global__ void zero_f32_kernel(float* __restrict__ p, int n) {
    int i = blockIdx.x * blockDim.x + threadIdx.x;
    if (i < n) p[i] = 0.f;
}

__global__ void cvt_f16_kernel(const float* __restrict__ s, _Float16* __restrict__ d, int n) {
    int i = blockIdx.x * blockDim.x + threadIdx.x;
    if (i < n) d[i] = (_Float16)s[i];
}

// n1_0 = relu(x @ Wn0^T), x:[N,16], Wn0:[64,16]
__global__ void n1_l0_kernel(const float* __restrict__ x, const float* __restrict__ Wn,
                             float* __restrict__ n1, int total) {
    int i = blockIdx.x * blockDim.x + threadIdx.x;
    if (i >= total) return;
    const int n = i >> 6, j = i & 63;
    const float* xr = x + n * 16;
    const float* wr = Wn + j * 16;
    float s = 0.f;
    #pragma unroll
    for (int k = 0; k < 16; ++k) s += xr[k] * wr[k];
    n1[i] = s > 0.f ? s : 0.f;
}

// layer-0 aggregation: agg0[dst[e]][j] += relu(ea[e]*We0[j])   (e1_0 never stored)
__global__ void e1agg_l0_kernel(const float* __restrict__ ea, const float* __restrict__ We,
                                const int* __restrict__ dst, float* __restrict__ agg,
                                int total) {
    int i = blockIdx.x * blockDim.x + threadIdx.x;
    if (i >= total) return;
    const int e = i >> 6, j = i & 63;
    float v = ea[e] * We[j];
    v = v > 0.f ? v : 0.f;
    atomicAdd(&agg[(size_t)dst[e] * 64 + j], v);
}

extern "C" void kernel_launch(void* const* d_in, const int* in_sizes, int n_in,
                              void* d_out, int out_size, void* d_ws, size_t ws_size,
                              hipStream_t stream) {
    (void)in_sizes; (void)n_in; (void)out_size; (void)ws_size;
    const int N = 50000, E = 1600000;

    const float* x      = (const float*)d_in[0];
    const float* ea     = (const float*)d_in[1];
    const int*   ei     = (const int*)d_in[2];
    const int*   src    = ei;
    const int*   dst    = ei + E;
    const float* l0_Wn  = (const float*)d_in[3];
    const float* l0_We  = (const float*)d_in[4];
    const float* l0_Wfn = (const float*)d_in[5];
    const float* l0_bfn = (const float*)d_in[6];
    const float* l0_Wfe = (const float*)d_in[7];
    const float* l0_bfe = (const float*)d_in[8];
    const float* l0_g   = (const float*)d_in[9];
    const float* l0_b   = (const float*)d_in[10];
    const float* l0_rm  = (const float*)d_in[11];
    const float* l0_rv  = (const float*)d_in[12];
    const float* l1_Wn  = (const float*)d_in[13];
    const float* l1_We  = (const float*)d_in[14];
    const float* l1_Wfn = (const float*)d_in[15];
    const float* l1_bfn = (const float*)d_in[16];
    // d_in[17]/d_in[18] (l1_Wfe/l1_bfe) are dead: layer-1 e2 is never returned.
    const float* l1_g   = (const float*)d_in[19];
    const float* l1_b   = (const float*)d_in[20];
    const float* l1_rm  = (const float*)d_in[21];
    const float* l1_rv  = (const float*)d_in[22];

    // ---- workspace layout ----
    char* ws = (char*)d_ws;
    size_t off = 0;
    auto walloc = [&](size_t bytes) -> void* {
        void* p = ws + off;
        off = (off + bytes + 255) & ~(size_t)255;
        return p;
    };
    float*     n1_0  = (float*)walloc((size_t)N * 64 * 4);
    float*     agg0  = (float*)walloc((size_t)N * 64 * 4);
    _Float16*  n2_0h = (_Float16*)walloc((size_t)N * 128 * 2);
    float*     n1_1  = (float*)walloc((size_t)N * 64 * 4);
    float*     agg1  = (float*)walloc((size_t)N * 64 * 4);
    _Float16*  Wfn0h = (_Float16*)walloc(128 * 128 * 2);
    _Float16*  Wfe0h = (_Float16*)walloc(128 * 128 * 2);
    _Float16*  Wn1h  = (_Float16*)walloc(64 * 128 * 2);
    _Float16*  We1h  = (_Float16*)walloc(64 * 128 * 2);
    _Float16*  Wfn1h = (_Float16*)walloc(128 * 128 * 2);
    _Float16*  e2_0h = (_Float16*)walloc((size_t)E * 128 * 2);

    const int nGridN = (N + 63) / 64;   // 64 rows per block (MTILES=4)
    const int nGridE = E / 64;          // exact

    // ---- zero aggregation buffers (workspace is not re-zeroed between calls) ----
    zero_f32_kernel<<<(N * 64 + 255) / 256, 256, 0, stream>>>(agg0, N * 64);
    zero_f32_kernel<<<(N * 64 + 255) / 256, 256, 0, stream>>>(agg1, N * 64);

    // ---- weight conversion to f16 (only the live matrices) ----
    cvt_f16_kernel<<<(128 * 128 + 255) / 256, 256, 0, stream>>>(l0_Wfn, Wfn0h, 128 * 128);
    cvt_f16_kernel<<<(128 * 128 + 255) / 256, 256, 0, stream>>>(l0_Wfe, Wfe0h, 128 * 128);
    cvt_f16_kernel<<<(64 * 128 + 255) / 256, 256, 0, stream>>>(l1_Wn, Wn1h, 64 * 128);
    cvt_f16_kernel<<<(64 * 128 + 255) / 256, 256, 0, stream>>>(l1_We, We1h, 64 * 128);
    cvt_f16_kernel<<<(128 * 128 + 255) / 256, 256, 0, stream>>>(l1_Wfn, Wfn1h, 128 * 128);

    // ---- layer 0 front half ----
    n1_l0_kernel<<<(N * 64 + 255) / 256, 256, 0, stream>>>(x, l0_Wn, n1_0, N * 64);
    e1agg_l0_kernel<<<(E * 64 + 255) / 256, 256, 0, stream>>>(ea, l0_We, dst, agg0, E * 64);

    // n2_0h = f16( BN0(relu(cat(agg0, n1_0) @ Wfn0^T + bfn0)) )
    gemm_wmma_kernel<STAGE_NODE_CAT, 128, 4, true, false, true, true>
        <<<nGridN, 256, 0, stream>>>(nullptr, n1_0, agg0, nullptr, nullptr, nullptr,
                                     Wfn0h, l0_bfn, l0_g, l0_b, l0_rm, l0_rv,
                                     n2_0h, nullptr, N);

    // e2_0h = f16( relu(cat(n1_0[src]+n1_0[dst], e1_0) @ Wfe0^T + bfe0) )
    gemm_wmma_kernel<STAGE_EDGE_CAT, 128, 4, false, false, true, true>
        <<<nGridE, 256, 0, stream>>>(nullptr, n1_0, ea, l0_We, src, dst,
                                     Wfe0h, l0_bfe, nullptr, nullptr, nullptr, nullptr,
                                     e2_0h, nullptr, E);

    // ---- layer 1 ----
    // n1_1 = relu(n2_0 @ Wn1^T)   (TDM-staged f16 A)
    gemm_wmma_kernel<STAGE_PLAIN_H, 64, 4, false, false, true, false>
        <<<nGridN, 128, 0, stream>>>(n2_0h, nullptr, nullptr, nullptr, nullptr, nullptr,
                                     Wn1h, nullptr, nullptr, nullptr, nullptr, nullptr,
                                     n1_1, nullptr, N);

    // e1_1 = relu(e2_0 @ We1^T); fused segment-sum into agg1, never stored
    gemm_wmma_kernel<STAGE_PLAIN_H, 64, 4, false, true, false, false>
        <<<nGridE, 128, 0, stream>>>(e2_0h, nullptr, nullptr, nullptr, nullptr, dst,
                                     We1h, nullptr, nullptr, nullptr, nullptr, nullptr,
                                     nullptr, agg1, E);

    // out = BN1(relu(cat(agg1, n1_1) @ Wfn1^T + bfn1))   (f32 output)
    gemm_wmma_kernel<STAGE_NODE_CAT, 128, 4, true, false, true, false>
        <<<nGridN, 256, 0, stream>>>(nullptr, n1_1, agg1, nullptr, nullptr, nullptr,
                                     Wfn1h, l1_bfn, l1_g, l1_b, l1_rm, l1_rv,
                                     (float*)d_out, nullptr, N);
}
